// RNN_20753281974858
// MI455X (gfx1250) — compile-verified, matrix operand order checked by reading, not run on hardware
//
#include <hip/hip_runtime.h>

// RNN: states[t] = tanh(X[t] @ W_ih + b + H_{t-1} @ W_hh)
// T=512, B=64, I=128, H=1024, fp32 in/out; bf16 WMMA with fp32 accumulate.

#define T_STEPS 512
#define BATCH   64
#define IN_DIM  128
#define HID     1024

typedef __attribute__((ext_vector_type(16))) __bf16 v16bf;
typedef __attribute__((ext_vector_type(8)))  __bf16 bf16x8;
typedef __attribute__((ext_vector_type(8)))  float  v8f;
typedef __attribute__((ext_vector_type(4)))  float  f32x4;

// Native fp32 -> bf16 convert (clang lowers to HW cvt on gfx1250).
__device__ __forceinline__ __bf16 f2bf(float f) { return (__bf16)f; }

// Build a 16-element bf16 WMMA operand from two contiguous 16-byte runs
// (K pattern per ISA: elements 0..7 -> K=kb..kb+7, 8..15 -> K=kb+16..kb+23).
// Union bit-cast lets the two global_load_b128 land directly in the
// fragment's 8-VGPR range with no per-element v_mov_b16 repacking.
__device__ __forceinline__ v16bf load_frag(const __bf16* __restrict__ base, int kb) {
    union { v16bf v; bf16x8 h[2]; } u;
    u.h[0] = *(const bf16x8*)(base + kb);
    u.h[1] = *(const bf16x8*)(base + kb + 16);
    return u.v;
}

// ---- one-time conversions -------------------------------------------------

// W_hh [K=1024][N=1024] fp32 -> WT [N=1024][K=1024] bf16 (transposed)
__global__ void conv_whh_kernel(const float* __restrict__ W,
                                __bf16* __restrict__ WT) {
    int idx = blockIdx.x * blockDim.x + threadIdx.x;   // over 1024*1024
    int k = idx >> 10;
    int n = idx & 1023;
    WT[n * HID + k] = f2bf(W[idx]);
}

// W_ih [K=128][N=1024] fp32 -> WiT [N=1024][K=128] bf16 (transposed)
__global__ void conv_wih_kernel(const float* __restrict__ W,
                                __bf16* __restrict__ WiT) {
    int idx = blockIdx.x * blockDim.x + threadIdx.x;   // over 128*1024
    int k = idx >> 10;
    int n = idx & 1023;
    WiT[n * IN_DIM + k] = f2bf(W[idx]);
}

// H0 [64][1024] fp32 -> bf16
__global__ void conv_h0_kernel(const float* __restrict__ H,
                               __bf16* __restrict__ Hb) {
    int idx = blockIdx.x * blockDim.x + threadIdx.x;   // over 64*1024
    Hb[idx] = f2bf(H[idx]);
}

// ---- input projection: Xp = X @ W_ih + b  (written into d_out states) -----
// One wave32 per 16x16 output tile. M = 32768 (T*B), N = 1024, K = 128.
__global__ __launch_bounds__(256) void input_proj_kernel(
        const float*  __restrict__ X,     // [32768][128] fp32
        const __bf16* __restrict__ WiT,   // [1024][128] bf16
        const float*  __restrict__ bias,  // [1024]
        float*        __restrict__ Xp) {  // [32768][1024] fp32 (d_out states)
    const int lane = threadIdx.x & 31;
    const int wave = threadIdx.x >> 5;
    const int tile = blockIdx.x * 8 + wave;        // 0 .. 131071
    const int nt    = tile & 63;                   // 64 N tiles
    const int mtile = tile >> 6;                   // 2048 M tiles
    const int m0 = mtile * 16;
    const int n0 = nt * 16;
    const int r    = lane & 15;
    const int half = lane >> 4;

    // C seeded with bias: C[M=i+8*half][N=r]
    const float bv = bias[n0 + r];
    v8f acc;
#pragma unroll
    for (int i = 0; i < 8; ++i) acc[i] = bv;

    const float*  arow = X   + (size_t)(m0 + r) * IN_DIM;
    const __bf16* brow = WiT + (size_t)(n0 + r) * IN_DIM;

#pragma unroll
    for (int k0 = 0; k0 < IN_DIM; k0 += 32) {
        const int kb = k0 + 8 * half;
        // A: fp32 row of X, converted to bf16 in registers (HW cvt)
        f32x4 x0 = *(const f32x4*)(arow + kb);
        f32x4 x1 = *(const f32x4*)(arow + kb + 4);
        f32x4 x2 = *(const f32x4*)(arow + kb + 16);
        f32x4 x3 = *(const f32x4*)(arow + kb + 20);
        union { v16bf v; __bf16 e[16]; } ua;
#pragma unroll
        for (int i = 0; i < 4; ++i) {
            ua.e[i]      = f2bf(x0[i]);
            ua.e[4 + i]  = f2bf(x1[i]);
            ua.e[8 + i]  = f2bf(x2[i]);
            ua.e[12 + i] = f2bf(x3[i]);
        }
        // B: bf16 row of W_ih^T, two contiguous 16B vectors
        v16bf b = load_frag(brow, kb);
        acc = __builtin_amdgcn_wmma_f32_16x16x32_bf16(
                false, ua.v, false, b, (short)0, acc, false, false);
    }

#pragma unroll
    for (int i = 0; i < 8; ++i)
        Xp[(size_t)(m0 + i + 8 * half) * HID + (n0 + r)] = acc[i];
}

// ---- one recurrence step: H_t = tanh(Xp_t + H_{t-1} @ W_hh) ---------------
// M = 64, N = 1024, K = 1024. 256 tiles -> 32 blocks x 8 waves, 32 WMMAs/wave.
__global__ __launch_bounds__(256) void rnn_step_kernel(
        const __bf16* __restrict__ Hprev,    // [64][1024] bf16
        const __bf16* __restrict__ WT,       // [1024][1024] bf16 (W_hh^T)
        float*        __restrict__ states_t, // [64][1024] fp32: in Xp_t, out H_t
        __bf16*       __restrict__ Hnext) {  // [64][1024] bf16
    const int lane = threadIdx.x & 31;
    const int wave = threadIdx.x >> 5;
    const int tile = blockIdx.x * 8 + wave;  // 0 .. 255
    const int mt = tile & 3;                 // 4 M tiles
    const int nt = tile >> 2;                // 64 N tiles
    const int m0 = mt * 16;
    const int n0 = nt * 16;
    const int r    = lane & 15;
    const int half = lane >> 4;

    // C seeded with Xp_t tile: C[M=i+8*half][N=r]
    v8f acc;
#pragma unroll
    for (int i = 0; i < 8; ++i)
        acc[i] = states_t[(m0 + i + 8 * half) * HID + (n0 + r)];

    const __bf16* arow = Hprev + (m0 + r) * HID;
    const __bf16* brow = WT    + (n0 + r) * HID;

#pragma unroll 8
    for (int k0 = 0; k0 < HID; k0 += 32) {
        const int kb = k0 + 8 * half;
        v16bf a = load_frag(arow, kb);
        v16bf b = load_frag(brow, kb);
        acc = __builtin_amdgcn_wmma_f32_16x16x32_bf16(
                false, a, false, b, (short)0, acc, false, false);
    }

#pragma unroll
    for (int i = 0; i < 8; ++i) {
        float h = tanhf(acc[i]);
        int idx = (m0 + i + 8 * half) * HID + (n0 + r);
        states_t[idx] = h;       // fp32 output (overwrites Xp_t in place)
        Hnext[idx]    = f2bf(h); // bf16 operand for next step
    }
}

// ---------------------------------------------------------------------------
extern "C" void kernel_launch(void* const* d_in, const int* in_sizes, int n_in,
                              void* d_out, int out_size, void* d_ws, size_t ws_size,
                              hipStream_t stream) {
    const float* X    = (const float*)d_in[0];   // [512,64,128]
    const float* W_ih = (const float*)d_in[1];   // [128,1024]
    const float* W_hh = (const float*)d_in[2];   // [1024,1024]
    const float* b_h  = (const float*)d_in[3];   // [1024]
    const float* H0   = (const float*)d_in[4];   // [64,1024]
    float* out = (float*)d_out;                  // states [512*64*1024] ++ Hfinal [64*1024]

    // workspace layout (~2.625 MB): W_hh^T bf16 | W_ih^T bf16 | H ping | H pong
    char* ws = (char*)d_ws;
    __bf16* WT  = (__bf16*)(ws);
    __bf16* WiT = (__bf16*)(ws + (size_t)HID * HID * 2);
    __bf16* Hb0 = (__bf16*)(ws + (size_t)HID * HID * 2 + (size_t)HID * IN_DIM * 2);
    __bf16* Hb1 = Hb0 + (size_t)BATCH * HID;
    __bf16* Hbuf[2] = {Hb0, Hb1};

    // one-time conversions (operands stay L2-resident afterwards)
    conv_whh_kernel<<<(HID * HID) / 256, 256, 0, stream>>>(W_hh, WT);
    conv_wih_kernel<<<(IN_DIM * HID) / 256, 256, 0, stream>>>(W_ih, WiT);
    conv_h0_kernel<<<(BATCH * HID) / 256, 256, 0, stream>>>(H0, Hbuf[0]);

    // batched input projection directly into the states region of d_out
    input_proj_kernel<<<(T_STEPS * BATCH / 16) * (HID / 16) / 8, 256, 0, stream>>>(
            X, WiT, b_h, out);

    // sequential recurrence: 512 dependent step kernels (graph-captured)
    for (int t = 0; t < T_STEPS; ++t) {
        rnn_step_kernel<<<32, 256, 0, stream>>>(
                Hbuf[t & 1], WT,
                out + (size_t)t * BATCH * HID,
                Hbuf[(t + 1) & 1]);
    }

    // H_final = states[T-1]
    hipMemcpyAsync(out + (size_t)T_STEPS * BATCH * HID,
                   out + (size_t)(T_STEPS - 1) * BATCH * HID,
                   (size_t)BATCH * HID * sizeof(float),
                   hipMemcpyDeviceToDevice, stream);
}
